// NSMCell_19988777796012
// MI455X (gfx1250) — matrix operand order repaired and checked by reading, not compiled, not run on hardware
//
#include <hip/hip_runtime.h>
#include <hip/hip_bf16.h>

typedef __bf16 bf16_t;
typedef __attribute__((ext_vector_type(16))) __bf16 v16bf;
typedef __attribute__((ext_vector_type(8)))  __bf16 v8bf;
typedef __attribute__((ext_vector_type(8)))  float  v8f;
typedef int v4i __attribute__((vector_size(16)));

#define GLOBAL_AS __attribute__((address_space(1)))
#define LDS_AS    __attribute__((address_space(3)))

#define B_    32
#define N_    64
#define P_    4
#define H_    300
#define HPAD  320          // edge K padded to 10*32
#define KKE   10           // edge k-steps of 32
#define KN    1200         // node K = P*H
#define KNPAD 1216         // 38*32
#define KKN   38           // node k-steps
#define CT    19           // output col tiles (19*16 = 304 >= 300)

// workspace layout (bytes)
#define REL_OFF   0
#define STATE_OFF 8192
#define WE_OFF    16384
#define WE_ELEMS  (CT * KKE * 32 * 16)        // 97280 bf16
#define WN_OFF    (WE_OFF + WE_ELEMS * 2)     // 210944 (32B aligned)
#define WN_ELEMS  (CT * KKN * 32 * 16)        // 369664 bf16

// ---------------------------------------------------------------------------
// Pack W_edge / W_node into WMMA B-fragment order and zero the accumulators.
// B fragment layout (16x16x32 bf16): lane l holds col = l%16;
// elements e=0..7 hold K = kk*32 + (l>=16)*8 + e, e=8..15 hold K = kk*32 + 16 + (l>=16)*8 + (e-8).
// ---------------------------------------------------------------------------
__global__ void pack_kernel(const float* __restrict__ wedge,
                            const float* __restrict__ wnode,
                            float* __restrict__ rel_acc,
                            float* __restrict__ state_raw,
                            bf16_t* __restrict__ we,
                            bf16_t* __restrict__ wn) {
  const int total = 4096 + WE_ELEMS + WN_ELEMS;
  for (int idx = blockIdx.x * blockDim.x + threadIdx.x; idx < total;
       idx += gridDim.x * blockDim.x) {
    if (idx < 2048) {
      rel_acc[idx] = 0.0f;
    } else if (idx < 4096) {
      state_raw[idx - 2048] = 0.0f;
    } else if (idx < 4096 + WE_ELEMS) {
      int j = idx - 4096;
      int e = j & 15, lane = (j >> 4) & 31, t = j >> 9;
      int kk = t % KKE, ct = t / KKE;
      int k = kk * 32 + ((lane & 16) ? 8 : 0) + (e & 7) + ((e & 8) ? 16 : 0);
      int col = ct * 16 + (lane & 15);
      we[j] = (k < H_ && col < H_) ? (bf16_t)wedge[k * H_ + col] : (bf16_t)0.0f;
    } else {
      int j = idx - 4096 - WE_ELEMS;
      int e = j & 15, lane = (j >> 4) & 31, t = j >> 9;
      int kk = t % KKN, ct = t / KKN;
      int k = kk * 32 + ((lane & 16) ? 8 : 0) + (e & 7) + ((e & 8) ? 16 : 0);
      int col = ct * 16 + (lane & 15);
      wn[j] = (k < KN && col < H_) ? (bf16_t)wnode[k * H_ + col] : (bf16_t)0.0f;
    }
  }
}

// A fragment (16x32 bf16, ISA layout): lane l holds row = l%16;
// elements 0..7 = K k0..k0+7 and 8..15 = K k0+16..k0+23, k0 = kk*32 + (l>=16)*8.
__device__ inline v16bf a_frag_from_bf16(const bf16_t* lds_row, int k0) {
  v8bf lo = *(const v8bf*)(lds_row + k0);
  v8bf hi = *(const v8bf*)(lds_row + k0 + 16);
  return __builtin_shufflevector(lo, hi, 0, 1, 2, 3, 4, 5, 6, 7, 8, 9, 10, 11,
                                 12, 13, 14, 15);
}

// Same fragment but converting from an fp32-staged LDS row (loop-invariant,
// so the cvts are hoisted out of the column-tile loop together with the frag).
__device__ inline v16bf a_frag_from_f32(const float* lds_row, int k0) {
  v16bf a;
#pragma unroll
  for (int e = 0; e < 8; ++e) {
    a[e]     = (bf16_t)lds_row[k0 + e];
    a[e + 8] = (bf16_t)lds_row[k0 + 16 + e];
  }
  return a;
}

// ---------------------------------------------------------------------------
// Edge path: rel_logits[b,m] += sum_n dist[b,n] * sum_k wrel[k] *
//            elu(instr[b,k] * (edge_attr[b,m,n,:] @ W_edge)[k])
// One wave per 16 consecutive (b,m,n) rows (all share b,m).
// A rows staged fp32 into LDS via CDNA5 async global->LDS DMA when available.
// ---------------------------------------------------------------------------
__global__ __launch_bounds__(32) void edge_kernel(
    const float* __restrict__ edge_attr, const float* __restrict__ instr,
    const float* __restrict__ dist, const float* __restrict__ wrel,
    const bf16_t* __restrict__ wpack, float* __restrict__ rel_acc) {
  __shared__ float Afp[16 * HPAD];  // 20 KB (fp32 staging)
  const int lane = threadIdx.x;
  const int rowBase = blockIdx.x * 16;
  const int b = rowBase >> 12;
  const int m = (rowBase >> 6) & 63;
  const int n0 = rowBase & 63;

  // 16 rows x 300 floats = 1200 16-byte chunks, 16B aligned (1200B row pitch).
#if __has_builtin(__builtin_amdgcn_global_load_async_to_lds_b128)
#pragma unroll 4
  for (int i = 0; i < 38; ++i) {
    int c = lane + 32 * i;
    if (c < 16 * 75) {
      int r = c / 75, q = c % 75;
      const float* g = edge_attr + (size_t)(rowBase + r) * H_ + q * 4;
      float* l = &Afp[r * HPAD + q * 4];
      __builtin_amdgcn_global_load_async_to_lds_b128(
          (GLOBAL_AS v4i*)g, (LDS_AS v4i*)l, 0, 0);
    }
  }
#else
#pragma unroll 4
  for (int i = 0; i < 38; ++i) {
    int c = lane + 32 * i;
    if (c < 16 * 75) {
      int r = c / 75, q = c % 75;
      *(float4*)&Afp[r * HPAD + q * 4] =
          *(const float4*)(edge_attr + (size_t)(rowBase + r) * H_ + q * 4);
    }
  }
#endif
  // Zero-pad K = 300..319 (plain LDS stores; disjoint from async region).
  if (lane < HPAD - H_) {
#pragma unroll
    for (int r = 0; r < 16; ++r) Afp[r * HPAD + H_ + lane] = 0.0f;
  }
#if __has_builtin(__builtin_amdgcn_s_wait_asynccnt)
  __builtin_amdgcn_s_wait_asynccnt(0);
#elif __has_builtin(__builtin_amdgcn_global_load_async_to_lds_b128)
  asm volatile("s_wait_asynccnt 0x0" ::: "memory");
#endif
  __syncthreads();

  const int hi = lane >> 4;
  const int row = lane & 15;
  const float* lds_row = &Afp[row * HPAD];
  float partial[8] = {0.f, 0.f, 0.f, 0.f, 0.f, 0.f, 0.f, 0.f};

  for (int ct = 0; ct < CT; ++ct) {
    const int col = ct * 16 + row;
    float iv = 0.0f, wv = 0.0f;
    if (col < H_) { iv = instr[b * H_ + col]; wv = wrel[col]; }

    v8f acc = {0.f, 0.f, 0.f, 0.f, 0.f, 0.f, 0.f, 0.f};
    const bf16_t* bp = wpack + ((size_t)(ct * KKE) * 32 + lane) * 16;
    // Pull next column-tile's B block toward the WGP caches.
    __builtin_prefetch(bp + (size_t)KKE * 512, 0, 3);
    // Software-pipelined B fragments: issue load of kk+1 before WMMA on kk.
    v16bf bm = *(const v16bf*)bp;
#pragma unroll
    for (int kk = 0; kk < KKE; ++kk) {
      v16bf bnext = bm;
      if (kk + 1 < KKE) bnext = *(const v16bf*)(bp + (size_t)(kk + 1) * 512);
      v16bf a = a_frag_from_f32(lds_row, kk * 32 + hi * 8);
      acc = __builtin_amdgcn_wmma_f32_16x16x32_bf16(false, a, false, bm,
                                                    (short)0, acc, false, false);
      bm = bnext;
    }
    // Fused elu + score-weighted accumulation.
    // C layout: lane l, slot v -> row = v + 8*(l>=16), col = ct*16 + l%16.
#pragma unroll
    for (int v = 0; v < 8; ++v) {
      float x = iv * acc[v];
      float e = x > 0.0f ? x : (__expf(x) - 1.0f);
      partial[v] += wv * e;
    }
  }

  // Sum cols across each 16-lane half (masks stay within the half).
#pragma unroll
  for (int mask = 1; mask <= 8; mask <<= 1)
#pragma unroll
    for (int v = 0; v < 8; ++v) partial[v] += __shfl_xor(partial[v], mask, 32);

  // Weight by distribution over the 16 n's this wave owns, reduce to scalar.
  float t = 0.0f;
#pragma unroll
  for (int v = 0; v < 8; ++v)
    t += partial[v] * dist[b * N_ + n0 + hi * 8 + v];
  t += __shfl_xor(t, 16, 32);
  if (lane == 0) atomicAdd(&rel_acc[b * N_ + m], t);
}

// ---------------------------------------------------------------------------
// Node path: state_raw[b,n] = sum_k wstate[k] *
//            elu(instr[b,k] * sum_{p,h} sims[b,p]*node_attr[b,n,p,h]*W[p,h,k])
// sims folded into the A operand during staging; K = P*H = 1200.
// ---------------------------------------------------------------------------
__global__ __launch_bounds__(32) void node_kernel(
    const float* __restrict__ node_attr, const float* __restrict__ instr,
    const float* __restrict__ sims, const float* __restrict__ wstate,
    const bf16_t* __restrict__ wpack, float* __restrict__ state_raw) {
  __shared__ bf16_t Alds[16 * KNPAD];  // 38 KB
  const int lane = threadIdx.x;
  const int rowBase = blockIdx.x * 16;
  const int b = rowBase >> 6;
  const int n0 = rowBase & 63;

#pragma unroll
  for (int p = 0; p < P_; ++p) {
    const float sp = sims[b * P_ + p];
    for (int r = 0; r < 16; ++r) {
      const float* src = node_attr + ((size_t)(rowBase + r) * P_ + p) * H_;
      for (int k = lane; k < H_; k += 32)
        Alds[r * KNPAD + p * H_ + k] = (bf16_t)(sp * src[k]);
    }
  }
  for (int r = 0; r < 16; ++r)
    if (lane < KNPAD - KN) Alds[r * KNPAD + KN + lane] = (bf16_t)0.0f;
  __syncthreads();

  const int hi = lane >> 4;
  const int row = lane & 15;
  const bf16_t* lds_row = &Alds[row * KNPAD];
  float partial[8] = {0.f, 0.f, 0.f, 0.f, 0.f, 0.f, 0.f, 0.f};

  for (int ct = 0; ct < CT; ++ct) {
    const int col = ct * 16 + row;
    float iv = 0.0f, wv = 0.0f;
    if (col < H_) { iv = instr[b * H_ + col]; wv = wstate[col]; }

    v8f acc = {0.f, 0.f, 0.f, 0.f, 0.f, 0.f, 0.f, 0.f};
    const bf16_t* bp = wpack + ((size_t)(ct * KKN) * 32 + lane) * 16;
    __builtin_prefetch(bp + (size_t)KKN * 512, 0, 3);
    v16bf bm = *(const v16bf*)bp;
#pragma unroll 2
    for (int kk = 0; kk < KKN; ++kk) {
      v16bf bnext = bm;
      if (kk + 1 < KKN) bnext = *(const v16bf*)(bp + (size_t)(kk + 1) * 512);
      v16bf a = a_frag_from_bf16(lds_row, kk * 32 + hi * 8);
      acc = __builtin_amdgcn_wmma_f32_16x16x32_bf16(false, a, false, bm,
                                                    (short)0, acc, false, false);
      bm = bnext;
    }
#pragma unroll
    for (int v = 0; v < 8; ++v) {
      float x = iv * acc[v];
      float e = x > 0.0f ? x : (__expf(x) - 1.0f);
      partial[v] += wv * e;
    }
  }

#pragma unroll
  for (int mask = 1; mask <= 8; mask <<= 1)
#pragma unroll
    for (int v = 0; v < 8; ++v) partial[v] += __shfl_xor(partial[v], mask, 32);

  if (lane == 0 || lane == 16) {
#pragma unroll
    for (int v = 0; v < 8; ++v)
      state_raw[b * N_ + n0 + hi * 8 + v] = partial[v];
  }
}

// ---------------------------------------------------------------------------
// Per-batch: add node_mask, softmax over the 64 entries, blend with r.
// ---------------------------------------------------------------------------
__global__ __launch_bounds__(32) void combine_kernel(
    const float* __restrict__ state_raw, const float* __restrict__ rel_acc,
    const float* __restrict__ node_mask, const float* __restrict__ relsim,
    float* __restrict__ out) {
  const int b = blockIdx.x, l = threadIdx.x;
  const float mk0 = node_mask[b * N_ + l];
  const float mk1 = node_mask[b * N_ + 32 + l];
  float s0 = state_raw[b * N_ + l] + mk0, s1 = state_raw[b * N_ + 32 + l] + mk1;
  float r0 = rel_acc[b * N_ + l] + mk0, r1 = rel_acc[b * N_ + 32 + l] + mk1;

  float ms = fmaxf(s0, s1), mr = fmaxf(r0, r1);
#pragma unroll
  for (int k = 1; k <= 16; k <<= 1) {
    ms = fmaxf(ms, __shfl_xor(ms, k, 32));
    mr = fmaxf(mr, __shfl_xor(mr, k, 32));
  }
  float es0 = __expf(s0 - ms), es1 = __expf(s1 - ms);
  float er0 = __expf(r0 - mr), er1 = __expf(r1 - mr);
  float ss = es0 + es1, sr = er0 + er1;
#pragma unroll
  for (int k = 1; k <= 16; k <<= 1) {
    ss += __shfl_xor(ss, k, 32);
    sr += __shfl_xor(sr, k, 32);
  }
  const float rr = relsim[b];
  out[b * N_ + l] = rr * (er0 / sr) + (1.0f - rr) * (es0 / ss);
  out[b * N_ + 32 + l] = rr * (er1 / sr) + (1.0f - rr) * (es1 / ss);
}

extern "C" void kernel_launch(void* const* d_in, const int* in_sizes, int n_in,
                              void* d_out, int out_size, void* d_ws,
                              size_t ws_size, hipStream_t stream) {
  const float* node_attr = (const float*)d_in[0];
  const float* edge_attr = (const float*)d_in[1];
  const float* instruction = (const float*)d_in[2];
  const float* distribution = (const float*)d_in[3];
  const float* sims = (const float*)d_in[4];
  const float* relsim = (const float*)d_in[5];
  const float* node_mask = (const float*)d_in[6];
  const float* wnode = (const float*)d_in[7];
  const float* wedge = (const float*)d_in[8];
  const float* wstate = (const float*)d_in[9];
  const float* wrel = (const float*)d_in[10];

  char* ws = (char*)d_ws;
  float* rel_acc = (float*)(ws + REL_OFF);
  float* state_raw = (float*)(ws + STATE_OFF);
  bf16_t* we = (bf16_t*)(ws + WE_OFF);
  bf16_t* wn = (bf16_t*)(ws + WN_OFF);

  hipLaunchKernelGGL(pack_kernel, dim3(512), dim3(256), 0, stream, wedge,
                     wnode, rel_acc, state_raw, we, wn);
  hipLaunchKernelGGL(node_kernel, dim3((B_ * N_) / 16), dim3(32), 0, stream,
                     node_attr, instruction, sims, wstate, wn, state_raw);
  hipLaunchKernelGGL(edge_kernel, dim3((B_ * N_ * N_) / 16), dim3(32), 0,
                     stream, edge_attr, instruction, distribution, wrel, we,
                     rel_acc);
  hipLaunchKernelGGL(combine_kernel, dim3(B_), dim3(32), 0, stream, state_raw,
                     rel_acc, node_mask, relsim, (float*)d_out);
}
